// LSTMTagger_81604378624743
// MI455X (gfx1250) — matrix operand order repaired and checked by reading, not compiled, not run on hardware
//
#include <hip/hip_runtime.h>
#include <math.h>

#define S_LEN 4096
#define L_CH  16
#define DW    256
#define DC    128
#define DH    512
#define T_TAGS 64

typedef __attribute__((ext_vector_type(16))) _Float16 v16h;
typedef __attribute__((ext_vector_type(8)))  float    v8f;

#define WMMA_F32_F16(a, b, c) \
  __builtin_amdgcn_wmma_f32_16x16x32_f16(false, (a), false, (b), (short)0, (c), false, false)

__device__ __forceinline__ float sigmoidf_(float x) { return 1.0f / (1.0f + expf(-x)); }

// ---- WMMA fragment loaders (layouts per CDNA5 ISA §7.12.2) ----

// A: 16x32 f16 (MxK), source row-major [.. x lda]. lane<16: row=lane, K {k0..k0+7, k0+16..k0+23};
// lane>=16: row=lane-16, K shifted by +8.
__device__ __forceinline__ v16h load_A_g(const _Float16* A, int lda, int m0, int k0, int lane) {
  const _Float16* p = A + (size_t)(m0 + (lane & 15)) * lda + k0 + ((lane & 16) ? 8 : 0);
  v16h a;
#pragma unroll
  for (int e = 0; e < 8; ++e) { a[e] = p[e]; a[e + 8] = p[e + 16]; }
  return a;
}

// B: 32x16 f16 (KxN) with B[k][n] = W[n][k], W row-major [N x ldw].
// lanes 0-15: n=lane, K {k0..k0+7, k0+16..k0+23} ; lanes 16-31: n=lane-16, K shifted by +8.
__device__ __forceinline__ v16h load_Bt_g(const _Float16* W, int ldw, int n0, int k0, int lane) {
  const _Float16* p = W + (size_t)(n0 + (lane & 15)) * ldw + k0 + ((lane & 16) ? 8 : 0);
  v16h b;
#pragma unroll
  for (int e = 0; e < 8; ++e) { b[e] = p[e]; b[e + 8] = p[e + 16]; }
  return b;
}

// ---------------- f32 -> f16 weight conversion ----------------
__global__ void cvt_f16_k(const float* __restrict__ in, _Float16* __restrict__ out, int n) {
  int i = blockIdx.x * 256 + threadIdx.x;
  if (i < n) out[i] = (_Float16)in[i];
}

// ---------------- char-level LSTM: 16 words per block, WMMA gates ----------------
// Block = 128 threads = 4 waves. Wave w owns j-range [32w, 32w+32) (two 16-wide n-tiles
// per gate type); gates/c/h for those j live entirely in registers.
__global__ __launch_bounds__(128) void char_lstm_k(
    const int* __restrict__ char_idxs, const int* __restrict__ char_lens,
    const float* __restrict__ char_emb,
    const _Float16* __restrict__ Wih, const _Float16* __restrict__ Whh,
    const float* __restrict__ bih, const float* __restrict__ bhh,
    float* __restrict__ charfeat) {
  __shared__ _Float16 xs[16][136];   // x_t tile, f16, padded rows
  __shared__ _Float16 hs[16][136];   // h tile, f16
  __shared__ int lens_s[16];

  const int tid = threadIdx.x, lane = tid & 31, wv = tid >> 5;
  const int s0 = blockIdx.x * 16;

  if (tid < 16) lens_s[tid] = char_lens[s0 + tid];
  {
    _Float16* hz = &hs[0][0];
    for (int i = tid; i < 16 * 136; i += 128) hz[i] = (_Float16)0.0f;
  }

  v8f cst[2];
#pragma unroll
  for (int jt = 0; jt < 2; ++jt) {
#pragma unroll
    for (int r = 0; r < 8; ++r) cst[jt][r] = 0.0f;
  }

  float bI[2], bF[2], bG[2], bO[2];
#pragma unroll
  for (int jt = 0; jt < 2; ++jt) {
    const int jl = (2 * wv + jt) * 16 + (lane & 15);
    bI[jt] = bih[0 * DC + jl] + bhh[0 * DC + jl];
    bF[jt] = bih[1 * DC + jl] + bhh[1 * DC + jl];
    bG[jt] = bih[2 * DC + jl] + bhh[2 * DC + jl];
    bO[jt] = bih[3 * DC + jl] + bhh[3 * DC + jl];
  }

  for (int t = 0; t < L_CH; ++t) {
    // gather char embeddings for this timestep -> xs (f16)
    {
      const int row = tid >> 3, seg = tid & 7;
      const int ci = char_idxs[(s0 + row) * L_CH + t];
      const float* src = char_emb + (size_t)ci * DC + seg * 16;
#pragma unroll
      for (int e = 0; e < 16; ++e) xs[row][seg * 16 + e] = (_Float16)src[e];
    }
    __syncthreads();

    v8f aI[2], aF[2], aG[2], aO[2];
#pragma unroll
    for (int jt = 0; jt < 2; ++jt) {
#pragma unroll
      for (int r = 0; r < 8; ++r) {
        aI[jt][r] = bI[jt]; aF[jt][r] = bF[jt]; aG[jt][r] = bG[jt]; aO[jt][r] = bO[jt];
      }
    }

#pragma unroll
    for (int kk = 0; kk < DC; kk += 32) {
      const v16h ax = load_A_g(&xs[0][0], 136, 0, kk, lane);
      const v16h ah = load_A_g(&hs[0][0], 136, 0, kk, lane);
#pragma unroll
      for (int jt = 0; jt < 2; ++jt) {
        const int nb = (2 * wv + jt) * 16;
        aI[jt] = WMMA_F32_F16(ax, load_Bt_g(Wih, DC, 0 * DC + nb, kk, lane), aI[jt]);
        aF[jt] = WMMA_F32_F16(ax, load_Bt_g(Wih, DC, 1 * DC + nb, kk, lane), aF[jt]);
        aG[jt] = WMMA_F32_F16(ax, load_Bt_g(Wih, DC, 2 * DC + nb, kk, lane), aG[jt]);
        aO[jt] = WMMA_F32_F16(ax, load_Bt_g(Wih, DC, 3 * DC + nb, kk, lane), aO[jt]);
        aI[jt] = WMMA_F32_F16(ah, load_Bt_g(Whh, DC, 0 * DC + nb, kk, lane), aI[jt]);
        aF[jt] = WMMA_F32_F16(ah, load_Bt_g(Whh, DC, 1 * DC + nb, kk, lane), aF[jt]);
        aG[jt] = WMMA_F32_F16(ah, load_Bt_g(Whh, DC, 2 * DC + nb, kk, lane), aG[jt]);
        aO[jt] = WMMA_F32_F16(ah, load_Bt_g(Whh, DC, 3 * DC + nb, kk, lane), aO[jt]);
      }
    }

    // pointwise cell math, fully register-resident (i/f/g/o share the D layout)
    float hnew[2][8];
#pragma unroll
    for (int jt = 0; jt < 2; ++jt) {
#pragma unroll
      for (int r = 0; r < 8; ++r) {
        const float ig = sigmoidf_(aI[jt][r]);
        const float fg = sigmoidf_(aF[jt][r]);
        const float gg = tanhf(aG[jt][r]);
        const float og = sigmoidf_(aO[jt][r]);
        const float c = fg * cst[jt][r] + ig * gg;
        cst[jt][r] = c;
        hnew[jt][r] = og * tanhf(c);
      }
    }
    __syncthreads();  // all waves done reading hs

    const int mh = (lane & 16) ? 8 : 0;
#pragma unroll
    for (int jt = 0; jt < 2; ++jt) {
      const int j = (2 * wv + jt) * 16 + (lane & 15);
#pragma unroll
      for (int r = 0; r < 8; ++r) {
        const int m = mh + r;
        hs[m][j] = (_Float16)hnew[jt][r];
        if (t == lens_s[m] - 1) charfeat[(size_t)(s0 + m) * DC + j] = hnew[jt][r];
      }
    }
  }
}

// ---------------- build wx = [word_emb | char_feat] as f16 ----------------
__global__ __launch_bounds__(128) void build_wx_k(
    const int* __restrict__ widx, const float* __restrict__ wemb,
    const float* __restrict__ cf, _Float16* __restrict__ wx) {
  const int s = blockIdx.x, tid = threadIdx.x;
  const int w = widx[s];
  for (int k = tid; k < DW; k += 128)
    wx[(size_t)s * (DW + DC) + k] = (_Float16)wemb[(size_t)w * DW + k];
  wx[(size_t)s * (DW + DC) + DW + tid] = (_Float16)cf[(size_t)s * DC + tid];
}

// ---------------- Xw = wx @ Wih_w^T + (bih_w + bhh_w), M=4096 K=384 N=2048 ----------------
__global__ __launch_bounds__(128) void gemm_xw_k(
    const _Float16* __restrict__ wx, const _Float16* __restrict__ Wih,
    const float* __restrict__ bih, const float* __restrict__ bhh,
    float* __restrict__ Xw) {
  const int lane = threadIdx.x & 31, wv = threadIdx.x >> 5;
  const int m0 = blockIdx.x * 16;
  const int n0 = (blockIdx.y * 4 + wv) * 16;
  const int n = n0 + (lane & 15);
  const float bias = bih[n] + bhh[n];
  v8f acc;
#pragma unroll
  for (int r = 0; r < 8; ++r) acc[r] = bias;
#pragma unroll
  for (int kk = 0; kk < (DW + DC); kk += 32)
    acc = WMMA_F32_F16(load_A_g(wx, DW + DC, m0, kk, lane),
                       load_Bt_g(Wih, DW + DC, n0, kk, lane), acc);
  const int mh = (lane & 16) ? 8 : 0;
#pragma unroll
  for (int r = 0; r < 8; ++r) Xw[(size_t)(m0 + mh + r) * (4 * DH) + n] = acc[r];
}

// ---------------- word-level LSTM: persistent, 32 blocks, grid-synced per step ----------------
// Block b owns h/c slice j in [16b, 16b+16); its 64 Whh rows (i/f/g/o) live in LDS (64KB f16).
__global__ __launch_bounds__(256) void word_lstm_k(
    const float* __restrict__ Xw, const _Float16* __restrict__ Whh,
    float* __restrict__ hbuf, _Float16* __restrict__ wout, int* __restrict__ ctr) {
  __shared__ _Float16 whh_s[64 * 512];
  __shared__ float hsrc[512];
  __shared__ float part[4][64];
  __shared__ float gates[64];
  __shared__ float cst[16];

  const int b = blockIdx.x, tid = threadIdx.x;

  for (int idx = tid; idx < 64 * 512; idx += 256) {
    const int r = idx >> 9, k = idx & 511;
    const int gt = r >> 4, j = r & 15;
    const int G = gt * DH + b * 16 + j;
    whh_s[idx] = Whh[(size_t)G * DH + k];
  }
  if (tid < 16) cst[tid] = 0.0f;
  __syncthreads();

  const int r = tid & 63, q = tid >> 6;

  for (int t = 0; t < S_LEN; ++t) {
    hsrc[tid] = hbuf[(t & 1) * DH + tid];
    hsrc[tid + 256] = hbuf[(t & 1) * DH + tid + 256];
    __syncthreads();

    const _Float16* wr = whh_s + r * 512 + q * 128;
    const float* hh = hsrc + q * 128;
    float sum = 0.0f;
#pragma unroll 8
    for (int k = 0; k < 128; ++k) sum = fmaf((float)wr[k], hh[k], sum);
    part[q][r] = sum;
    __syncthreads();

    if (tid < 64) {
      const int gt = tid >> 4, j = tid & 15;
      const int G = gt * DH + b * 16 + j;
      gates[tid] = part[0][tid] + part[1][tid] + part[2][tid] + part[3][tid] +
                   Xw[(size_t)t * (4 * DH) + G];
    }
    __syncthreads();

    if (tid < 16) {
      const int j = tid;
      const float ig = sigmoidf_(gates[j]);
      const float fg = sigmoidf_(gates[16 + j]);
      const float gg = tanhf(gates[32 + j]);
      const float og = sigmoidf_(gates[48 + j]);
      const float c = fg * cst[j] + ig * gg;
      cst[j] = c;
      const float h = og * tanhf(c);
      hbuf[((t + 1) & 1) * DH + b * 16 + j] = h;                  // double-buffered h
      wout[(size_t)t * DH + b * 16 + j] = (_Float16)h;            // for tag GEMM
    }
    __threadfence();
    __syncthreads();
    if (tid == 0) {
      __hip_atomic_fetch_add(ctr, 1, __ATOMIC_RELEASE, __HIP_MEMORY_SCOPE_AGENT);
      while (__hip_atomic_load(ctr, __ATOMIC_ACQUIRE, __HIP_MEMORY_SCOPE_AGENT) <
             32 * (t + 1)) {
        __builtin_amdgcn_s_sleep(2);
      }
    }
    __syncthreads();
  }
}

// ---------------- tag_space = wout @ W_tag^T + b_tag, M=4096 K=512 N=64 ----------------
__global__ __launch_bounds__(128) void gemm_tag_k(
    const _Float16* __restrict__ wout, const _Float16* __restrict__ Wt,
    const float* __restrict__ bt, float* __restrict__ tag) {
  const int lane = threadIdx.x & 31, wv = threadIdx.x >> 5;
  const int m0 = blockIdx.x * 16;
  const int n0 = wv * 16;
  const float bias = bt[n0 + (lane & 15)];
  v8f acc;
#pragma unroll
  for (int r = 0; r < 8; ++r) acc[r] = bias;
#pragma unroll
  for (int kk = 0; kk < DH; kk += 32)
    acc = WMMA_F32_F16(load_A_g(wout, DH, m0, kk, lane),
                       load_Bt_g(Wt, DH, n0, kk, lane), acc);
  const int mh = (lane & 16) ? 8 : 0;
  const int n = n0 + (lane & 15);
#pragma unroll
  for (int r = 0; r < 8; ++r) tag[(size_t)(m0 + mh + r) * T_TAGS + n] = acc[r];
}

// ---------------- log_softmax over axis 0 (one block per tag column) ----------------
__global__ __launch_bounds__(256) void logsoftmax_k(const float* __restrict__ tag,
                                                    float* __restrict__ out) {
  __shared__ float red[256];
  const int t = blockIdx.x, tid = threadIdx.x;
  float m = -3.4e38f;
  for (int s = tid; s < S_LEN; s += 256) m = fmaxf(m, tag[(size_t)s * T_TAGS + t]);
  red[tid] = m;
  __syncthreads();
  for (int o = 128; o > 0; o >>= 1) {
    if (tid < o) red[tid] = fmaxf(red[tid], red[tid + o]);
    __syncthreads();
  }
  const float mx = red[0];
  __syncthreads();
  float sum = 0.0f;
  for (int s = tid; s < S_LEN; s += 256) sum += expf(tag[(size_t)s * T_TAGS + t] - mx);
  red[tid] = sum;
  __syncthreads();
  for (int o = 128; o > 0; o >>= 1) {
    if (tid < o) red[tid] += red[tid + o];
    __syncthreads();
  }
  const float ls = logf(red[0]);
  for (int s = tid; s < S_LEN; s += 256)
    out[(size_t)s * T_TAGS + t] = tag[(size_t)s * T_TAGS + t] - mx - ls;
}

// ---------------- host launcher ----------------
extern "C" void kernel_launch(void* const* d_in, const int* in_sizes, int n_in,
                              void* d_out, int out_size, void* d_ws, size_t ws_size,
                              hipStream_t stream) {
  (void)in_sizes; (void)n_in; (void)out_size; (void)ws_size;

  const int*   word_idxs = (const int*)d_in[0];
  const int*   char_idxs = (const int*)d_in[1];
  const int*   char_lens = (const int*)d_in[2];
  const float* word_emb  = (const float*)d_in[3];
  const float* char_emb  = (const float*)d_in[4];
  const float* Wih_c     = (const float*)d_in[5];
  const float* Whh_c     = (const float*)d_in[6];
  const float* bih_c     = (const float*)d_in[7];
  const float* bhh_c     = (const float*)d_in[8];
  const float* Wih_w     = (const float*)d_in[9];
  const float* Whh_w     = (const float*)d_in[10];
  const float* bih_w     = (const float*)d_in[11];
  const float* bhh_w     = (const float*)d_in[12];
  const float* W_tag     = (const float*)d_in[13];
  const float* b_tag     = (const float*)d_in[14];
  float* out = (float*)d_out;

  char* ws = (char*)d_ws;
  size_t o = 0;
  auto take = [&](size_t bytes) -> char* {
    size_t r = (o + 255) & ~(size_t)255;
    o = r + bytes;
    return ws + r;
  };
  int*      ctr    = (int*)take(256);
  float*    hbuf   = (float*)take((size_t)2 * DH * sizeof(float));
  _Float16* Wihc_h = (_Float16*)take((size_t)4 * DC * DC * 2);
  _Float16* Whhc_h = (_Float16*)take((size_t)4 * DC * DC * 2);
  _Float16* Wihw_h = (_Float16*)take((size_t)4 * DH * (DW + DC) * 2);
  _Float16* Whhw_h = (_Float16*)take((size_t)4 * DH * DH * 2);
  _Float16* Wtag_h = (_Float16*)take((size_t)T_TAGS * DH * 2);
  float*    cfeat  = (float*)take((size_t)S_LEN * DC * 4);
  _Float16* wx     = (_Float16*)take((size_t)S_LEN * (DW + DC) * 2);
  float*    Xw     = (float*)take((size_t)S_LEN * 4 * DH * 4);
  _Float16* wout   = (_Float16*)take((size_t)S_LEN * DH * 2);
  float*    tag    = (float*)take((size_t)S_LEN * T_TAGS * 4);

  // zero barrier counter + h double-buffer (contiguous at workspace start)
  hipMemsetAsync(d_ws, 0, 256 + (size_t)2 * DH * sizeof(float) + 256, stream);

  cvt_f16_k<<<(4 * DC * DC + 255) / 256, 256, 0, stream>>>(Wih_c, Wihc_h, 4 * DC * DC);
  cvt_f16_k<<<(4 * DC * DC + 255) / 256, 256, 0, stream>>>(Whh_c, Whhc_h, 4 * DC * DC);
  cvt_f16_k<<<(4 * DH * (DW + DC) + 255) / 256, 256, 0, stream>>>(Wih_w, Wihw_h,
                                                                  4 * DH * (DW + DC));
  cvt_f16_k<<<(4 * DH * DH + 255) / 256, 256, 0, stream>>>(Whh_w, Whhw_h, 4 * DH * DH);
  cvt_f16_k<<<(T_TAGS * DH + 255) / 256, 256, 0, stream>>>(W_tag, Wtag_h, T_TAGS * DH);

  char_lstm_k<<<S_LEN / 16, 128, 0, stream>>>(char_idxs, char_lens, char_emb,
                                              Wihc_h, Whhc_h, bih_c, bhh_c, cfeat);
  build_wx_k<<<S_LEN, 128, 0, stream>>>(word_idxs, word_emb, cfeat, wx);
  gemm_xw_k<<<dim3(S_LEN / 16, (4 * DH) / 64), 128, 0, stream>>>(wx, Wihw_h, bih_w,
                                                                 bhh_w, Xw);
  word_lstm_k<<<32, 256, 0, stream>>>(Xw, Whhw_h, hbuf, wout, ctr);
  gemm_tag_k<<<S_LEN / 16, 128, 0, stream>>>(wout, Wtag_h, b_tag, tag);
  logsoftmax_k<<<T_TAGS, 256, 0, stream>>>(tag, out);
}
